// MHA_66984309948830
// MI455X (gfx1250) — matrix operand order repaired and checked by reading, not compiled, not run on hardware
//
#include <hip/hip_runtime.h>

typedef __bf16 bf16;
typedef __attribute__((ext_vector_type(16))) __bf16 v16bf;
typedef __attribute__((ext_vector_type(8)))  float  v8f;

constexpr int kH   = 12;
constexpr int kE   = 768;
constexpr int kE3  = 2304;
constexpr int kD   = 64;
constexpr int kL   = 2048;
constexpr int kTok = 4096;          // B * L
constexpr float kLoraScale = 8.0f;  // 32.0 / 4.0

// ---------------- CDNA5 async global->LDS staging (ISA 15.18.3, ASYNCcnt) ----------------
__device__ inline unsigned lds_off(const void* p) { return (unsigned)(size_t)p; }
// GLOBAL_LOAD_ASYNC_TO_LDS_B128, GV mode: vdst = LDS byte address, vaddr = 64-bit global addr
__device__ inline void async_ld_b128(void* lds_dst, const void* gsrc) {
  asm volatile("global_load_async_to_lds_b128 %0, %1, off"
               :: "v"(lds_off(lds_dst)), "v"(gsrc) : "memory");
}
__device__ inline void wait_asynccnt0() {
  asm volatile("s_wait_asynccnt 0x0" ::: "memory");
}

// ---------------- fragment helpers (layouts per CDNA5 ISA) ----------------
// A 16x32 bf16 frag: lane m = lane%16; element e: K = lh*16 + e (contiguous)
__device__ inline v16bf frag_a_lds(const bf16* rowp, int lh) {
  union { uint4 q[2]; v16bf v; } u;
  u.q[0] = *(const uint4*)(rowp + lh * 16);
  u.q[1] = *(const uint4*)(rowp + lh * 16 + 8);
  return u.v;
}
// B 32x16 bf16 frag: lane n = lane%16; element e: K = lh*16 + e (32 contiguous bf16)
__device__ inline v16bf frag_b_lds(const bf16* p) {
  union { uint4 q[2]; v16bf v; } u;
  u.q[0] = *(const uint4*)(p);
  u.q[1] = *(const uint4*)(p + 8);
  return u.v;
}
__device__ inline v8f wmma_bf16(v16bf a, v16bf b, v8f c) {
  return __builtin_amdgcn_wmma_f32_16x16x32_bf16(false, a, false, b, (short)0, c, false, false);
}
__device__ inline v8f vzero8() { v8f z = {0.f,0.f,0.f,0.f,0.f,0.f,0.f,0.f}; return z; }

__device__ inline uint4 pack8_bf16(v8f v, float scale) {
  union { uint4 q; bf16 h[8]; } u;
#pragma unroll
  for (int j = 0; j < 8; ++j) u.h[j] = (bf16)(v[j] * scale);
  return u.q;
}
__device__ inline uint4 pack8_bf16f(const float* p) {
  union { uint4 q; bf16 h[8]; } u;
#pragma unroll
  for (int j = 0; j < 8; ++j) u.h[j] = (bf16)p[j];
  return u.q;
}

// ---------------- prep: fused LoRA weight, bf16 convert, TRANSPOSED store [o][e] ----------------
__global__ void prep_w_kernel(const float* __restrict__ w, const float* __restrict__ A,
                              const float* __restrict__ Bm, bf16* __restrict__ outT, int ncols) {
  int idx = blockIdx.x * 256 + threadIdx.x;            // idx = o*kE + e (output-major)
  if (idx >= kE * ncols) return;
  int o = idx / kE, e = idx - o * kE;
  float acc = w[(size_t)e * ncols + o];
#pragma unroll
  for (int r = 0; r < 4; ++r)
    acc += kLoraScale * Bm[e * 4 + r] * A[r * ncols + o];
  outT[idx] = (bf16)acc;                               // [o][e] layout
}

__global__ void cvt_x_kernel(const float* __restrict__ x, bf16* __restrict__ xb) {
  int idx = blockIdx.x * 256 + threadIdx.x;
  xb[idx] = (bf16)x[idx];
}

// ==== async-staged double-buffered 128x128 GEMM body, 8 waves, 2 M-strips x 4 N-subtiles ====
// A: rows = tokens [m][k-major], B: rows = outputs [o][k-major] (pre-transposed weights)
// Staging: global_load_async_to_lds_b128 (no VGPRs); per K-step per wave:
// 2 A-frag + 4 B-frag ds loads feed 8 WMMAs (each B reused x2).
#define GEMM_BODY(APTR, BPTR)                                                          \
  __shared__ bf16 Ab[2][128 * 32];                                                     \
  __shared__ bf16 Bb[2][128 * 32];                                                     \
  const int tid = threadIdx.x, wv = tid >> 5, lane = tid & 31;                         \
  const int lh = lane >> 4, ln = lane & 15;                                            \
  const int mq = wv >> 1, nh = wv & 1;   /* wave = (M quad, N half) */                 \
  const int m0 = blockIdx.x * 128, n0 = blockIdx.y * 128;                              \
  const int rbase = tid >> 2, cc = (tid & 3) * 8;                                      \
  v8f acc[2][4];                                                                       \
  _Pragma("unroll") for (int s = 0; s < 2; ++s)                                        \
    _Pragma("unroll") for (int t = 0; t < 4; ++t) acc[s][t] = vzero8();                \
  _Pragma("unroll") for (int i = 0; i < 2; ++i) {                                      \
    async_ld_b128(Ab[0] + (i * 64 + rbase) * 32 + cc,                                  \
                  APTR + (size_t)(m0 + i * 64 + rbase) * kE + cc);                     \
    async_ld_b128(Bb[0] + (i * 64 + rbase) * 32 + cc,                                  \
                  BPTR + (size_t)(n0 + i * 64 + rbase) * kE + cc);                     \
  }                                                                                    \
  wait_asynccnt0();                                                                    \
  __syncthreads();                                                                     \
  for (int k0 = 0; k0 < kE; k0 += 32) {                                                \
    const int cur = (k0 >> 5) & 1, nxt = cur ^ 1;                                      \
    const bool more = (k0 + 32 < kE);                                                  \
    if (more) {   /* background DMA of next tile while WMMAs run */                    \
      _Pragma("unroll") for (int i = 0; i < 2; ++i) {                                  \
        async_ld_b128(Ab[nxt] + (i * 64 + rbase) * 32 + cc,                            \
                      APTR + (size_t)(m0 + i * 64 + rbase) * kE + k0 + 32 + cc);       \
        async_ld_b128(Bb[nxt] + (i * 64 + rbase) * 32 + cc,                            \
                      BPTR + (size_t)(n0 + i * 64 + rbase) * kE + k0 + 32 + cc);       \
      }                                                                                \
    }                                                                                  \
    v16bf a0 = frag_a_lds(Ab[cur] + (mq * 32 + ln) * 32, lh);                          \
    v16bf a1 = frag_a_lds(Ab[cur] + (mq * 32 + 16 + ln) * 32, lh);                     \
    v16bf b[4];                                                                        \
    _Pragma("unroll") for (int t = 0; t < 4; ++t)                                      \
      b[t] = frag_b_lds(Bb[cur] + (nh * 64 + t * 16 + ln) * 32 + lh * 16);             \
    _Pragma("unroll") for (int t = 0; t < 4; ++t) {                                    \
      acc[0][t] = wmma_bf16(a0, b[t], acc[0][t]);                                      \
      acc[1][t] = wmma_bf16(a1, b[t], acc[1][t]);                                      \
    }                                                                                  \
    if (more) wait_asynccnt0();                                                        \
    __syncthreads();                                                                   \
  }

// ---------------- QKV GEMM: C[4096,2304] = xb @ w1^T + b_in -> Q/K/Vt ----------------
__launch_bounds__(256, 4)
__global__ void qkv_gemm_kernel(const bf16* __restrict__ xb, const bf16* __restrict__ w1T,
                                const float* __restrict__ b_in,
                                bf16* __restrict__ qw, bf16* __restrict__ kw, bf16* __restrict__ vtw) {
  GEMM_BODY(xb, w1T)
  // epilogue: bias, bf16, scatter into Q[bh,l,d], K[bh,l,d], Vt[bh,d,l]
#pragma unroll
  for (int s = 0; s < 2; ++s) {
#pragma unroll
    for (int t = 0; t < 4; ++t) {
      int o = n0 + nh * 64 + t * 16 + ln;
      float bias = b_in[o];
      if (o < 2 * kE) {              // Q / K: per-element stores (l-major dest)
#pragma unroll
        for (int r = 0; r < 8; ++r) {
          int tok = m0 + mq * 32 + s * 16 + r + 8 * lh;
          int bb = tok >> 11, l = tok & 2047;
          float v = acc[s][t][r] + bias;
          int oo = (o < kE) ? o : (o - kE);
          int h = oo >> 6, d = oo & 63;
          bf16* dst = (o < kE) ? qw : kw;
          dst[(((size_t)(bb * kH + h)) * kL + l) * kD + d] = (bf16)v;
        }
      } else {                       // V^T: tokens contiguous -> packed b128 store
        int o2 = o - 2 * kE, h = o2 >> 6, d = o2 & 63;
        int tok0 = m0 + mq * 32 + s * 16 + 8 * lh;
        int bb = tok0 >> 11, l0 = tok0 & 2047;
        float tmp[8];
#pragma unroll
        for (int r = 0; r < 8; ++r) tmp[r] = acc[s][t][r] + bias;
        *(uint4*)(vtw + (((size_t)(bb * kH + h)) * kD + d) * kL + l0) = pack8_bf16f(tmp);
      }
    }
  }
}

// ---------------- output GEMM: out[4096,768] = aw @ w2^T + b_out (fp32 out) ----------------
__launch_bounds__(256, 4)
__global__ void out_gemm_kernel(const bf16* __restrict__ aw, const bf16* __restrict__ w2T,
                                const float* __restrict__ b_out, float* __restrict__ outp) {
  GEMM_BODY(aw, w2T)
#pragma unroll
  for (int s = 0; s < 2; ++s) {
#pragma unroll
    for (int t = 0; t < 4; ++t) {
      int o = n0 + nh * 64 + t * 16 + ln;
      float bias = b_out[o];
#pragma unroll
      for (int r = 0; r < 8; ++r) {
        int tok = m0 + mq * 32 + s * 16 + r + 8 * lh;
        outp[(size_t)tok * kE + o] = acc[s][t][r] + bias;
      }
    }
  }
}

// ---------------- flash attention (swapped operands: S' = K x Q^T, O' = V^T x P) ----------------
__launch_bounds__(128, 4)
__global__ void attn_kernel(const bf16* __restrict__ qw, const bf16* __restrict__ kw,
                            const bf16* __restrict__ vtw, bf16* __restrict__ aw) {
  __shared__ bf16 Qlds[64 * 64];        // [q][d]
  __shared__ bf16 Kb[2][64 * 64];       // [key][d], double-buffered
  __shared__ bf16 Vb[2][64 * 64];       // [d][key], double-buffered
  __shared__ bf16 Plds[4 * 16 * 64];    // per-wave P strip [q][key]
  const int tid = threadIdx.x, wv = tid >> 5, lane = tid & 31;
  const int lh = lane >> 4, ln = lane & 15;
  const int bh = blockIdx.y, qt = blockIdx.x, q0 = qt * 64;
  const bf16* qbase = qw + (size_t)bh * kL * kD;
  const bf16* kbase = kw + (size_t)bh * kL * kD;
  const bf16* vbase = vtw + (size_t)bh * kD * kL;
  // staging geometry: 512 x 16B chunks per 64x64 tile, 4 per thread
  const int srow[4] = { (tid * 4 + 0) >> 3, (tid * 4 + 1) >> 3, (tid * 4 + 2) >> 3, (tid * 4 + 3) >> 3 };
  const int scol[4] = { ((tid * 4 + 0) & 7) * 8, ((tid * 4 + 1) & 7) * 8,
                        ((tid * 4 + 2) & 7) * 8, ((tid * 4 + 3) & 7) * 8 };

#pragma unroll
  for (int i = 0; i < 4; ++i) {   // async stage Q tile + K/Vt tile 0
    async_ld_b128(Qlds + srow[i] * 64 + scol[i], qbase + (size_t)(q0 + srow[i]) * kD + scol[i]);
    async_ld_b128(Kb[0] + srow[i] * 64 + scol[i], kbase + (size_t)srow[i] * kD + scol[i]);
    async_ld_b128(Vb[0] + srow[i] * 64 + scol[i], vbase + (size_t)srow[i] * kL + scol[i]);
  }
  wait_asynccnt0();
  __syncthreads();
  // Q as B operand: lane n = q (wave's 16-row strip), contiguous d
  v16bf qb[2];
  qb[0] = frag_b_lds(Qlds + (wv * 16 + ln) * 64 + 0  + lh * 16);
  qb[1] = frag_b_lds(Qlds + (wv * 16 + ln) * 64 + 32 + lh * 16);

  v8f O[4];                    // O' tiles: M = d (td*16 + r + 8lh), N = q = ln
#pragma unroll
  for (int t = 0; t < 4; ++t) O[t] = vzero8();
  float mcur = -3.0e38f, lcur = 0.f;   // per-lane = per q-row stats
  const int qrow = q0 + wv * 16 + ln;

  const int nkt = qt + 1;
  for (int it = 0; it < nkt; ++it) {
    const int cur = it & 1, nxt = cur ^ 1;
    const bool more = (it + 1 < nkt);
    if (more) {                 // background DMA of next K/Vt tile
      int k0n = (it + 1) * 64;
#pragma unroll
      for (int i = 0; i < 4; ++i) {
        async_ld_b128(Kb[nxt] + srow[i] * 64 + scol[i],
                      kbase + (size_t)(k0n + srow[i]) * kD + scol[i]);
        async_ld_b128(Vb[nxt] + srow[i] * 64 + scol[i],
                      vbase + (size_t)srow[i] * kL + k0n + scol[i]);
      }
    }

    v8f S[4];                  // S' tiles: M = key, N = q
#pragma unroll
    for (int t = 0; t < 4; ++t) S[t] = vzero8();
#pragma unroll
    for (int ks = 0; ks < 2; ++ks) {
      v16bf ka[4];
#pragma unroll
      for (int t = 0; t < 4; ++t)
        ka[t] = frag_a_lds(Kb[cur] + (t * 16 + ln) * 64 + ks * 32, lh);
#pragma unroll
      for (int t = 0; t < 4; ++t) S[t] = wmma_bf16(ka[t], qb[ks], S[t]);
    }

    if (it == qt) {            // causal mask on diagonal tile
      int k0 = it * 64;
#pragma unroll
      for (int t = 0; t < 4; ++t)
#pragma unroll
        for (int r = 0; r < 8; ++r)
          if (k0 + t * 16 + r + 8 * lh > qrow) S[t][r] = -1.0e9f;
    }

    // online softmax: each lane owns q-row `qrow`; reduce in-lane + one xor-16
    float rm = -3.0e38f;
#pragma unroll
    for (int t = 0; t < 4; ++t)
#pragma unroll
      for (int r = 0; r < 8; ++r) rm = fmaxf(rm, S[t][r]);
    rm = fmaxf(rm, __shfl_xor(rm, 16));
    float mnew = fmaxf(mcur, rm);
    float sc = __expf(mcur - mnew);
    float rs = 0.f;
    bf16* prow = Plds + wv * (16 * 64) + ln * 64;
#pragma unroll
    for (int t = 0; t < 4; ++t) {
      float p[8];
#pragma unroll
      for (int r = 0; r < 8; ++r) { p[r] = __expf(S[t][r] - mnew); rs += p[r]; }
      *(uint4*)(prow + t * 16 + 8 * lh) = pack8_bf16f(p);  // keys contiguous
    }
    rs += __shfl_xor(rs, 16);
    lcur = lcur * sc + rs;
    mcur = mnew;
#pragma unroll
    for (int td = 0; td < 4; ++td)
#pragma unroll
      for (int j = 0; j < 8; ++j) O[td][j] *= sc;

    // O' += V^T x P
#pragma unroll
    for (int ks = 0; ks < 2; ++ks) {
      v16bf pb = frag_b_lds(prow + ks * 32 + lh * 16);
      v16bf va[4];
#pragma unroll
      for (int td = 0; td < 4; ++td)
        va[td] = frag_a_lds(Vb[cur] + (td * 16 + ln) * 64 + ks * 32, lh);
#pragma unroll
      for (int td = 0; td < 4; ++td) O[td] = wmma_bf16(va[td], pb, O[td]);
    }

    if (more) wait_asynccnt0();
    __syncthreads();            // single barrier per key tile
  }
  // write: O'[d][q] -> aw[b, qrow, h*64 + d]; d contiguous per lane -> b128 stores
  const int h = bh % kH, bb = bh / kH;
  const float rinv = 1.0f / lcur;
  bf16* orow = aw + ((size_t)(bb * kL + qrow)) * kE + h * 64;
#pragma unroll
  for (int td = 0; td < 4; ++td)
    *(uint4*)(orow + td * 16 + 8 * lh) = pack8_bf16(O[td], rinv);
}

// ---------------- launcher ----------------
extern "C" void kernel_launch(void* const* d_in, const int* in_sizes, int n_in,
                              void* d_out, int out_size, void* d_ws, size_t ws_size,
                              hipStream_t stream) {
  const float* x     = (const float*)d_in[0];
  const float* w_in  = (const float*)d_in[1];
  const float* b_in  = (const float*)d_in[2];
  const float* A_in  = (const float*)d_in[3];
  const float* B_in  = (const float*)d_in[4];
  const float* w_out = (const float*)d_in[5];
  const float* b_out = (const float*)d_in[6];
  const float* A_out = (const float*)d_in[7];
  const float* B_out = (const float*)d_in[8];
  float* outp = (float*)d_out;

  // workspace carve-up (~36.1 MB total, all bf16)
  char* ws = (char*)d_ws;
  bf16* xb   = (bf16*)ws; ws += (size_t)kTok * kE * 2;        // x in bf16
  bf16* w1T  = (bf16*)ws; ws += (size_t)kE * kE3 * 2;         // fused in-proj, [o][e]
  bf16* w2T  = (bf16*)ws; ws += (size_t)kE * kE * 2;          // fused out-proj, [o][e]
  bf16* qw   = (bf16*)ws; ws += (size_t)2 * kH * kL * kD * 2; // Q [b,h,l,d]
  bf16* kw   = (bf16*)ws; ws += (size_t)2 * kH * kL * kD * 2; // K [b,h,l,d]
  bf16* vtw  = (bf16*)ws; ws += (size_t)2 * kH * kD * kL * 2; // V^T [b,h,d,l]
  bf16* aw   = (bf16*)ws;                                     // attn out [b,l,E]

  prep_w_kernel<<<(kE * kE3 + 255) / 256, 256, 0, stream>>>(w_in, A_in, B_in, w1T, kE3);
  prep_w_kernel<<<(kE * kE + 255) / 256, 256, 0, stream>>>(w_out, A_out, B_out, w2T, kE);
  cvt_x_kernel<<<(kTok * kE) / 256, 256, 0, stream>>>(x, xb);
  qkv_gemm_kernel<<<dim3(kTok / 128, kE3 / 128), 256, 0, stream>>>(xb, w1T, b_in, qw, kw, vtw);
  attn_kernel<<<dim3(kL / 64, 2 * kH), 128, 0, stream>>>(qw, kw, vtw, aw);
  out_gemm_kernel<<<dim3(kTok / 128, kE / 128), 256, 0, stream>>>(aw, w2T, b_out, outp);
}
